// AttentionBlock_45621142618525
// MI455X (gfx1250) — matrix operand order repaired and checked by reading, not compiled, not run on hardware
//
#include <hip/hip_runtime.h>

// Problem constants
#define Bz 8
#define Cc 1024
#define Ll 4096
#define Hh 16
#define CHh 64
#define CPG 32          // channels per group (1024 / 32 groups)
#define L3C 3072

typedef __attribute__((ext_vector_type(16))) __bf16        v16bf;
typedef __attribute__((ext_vector_type(8)))  float         v8f;
typedef __attribute__((ext_vector_type(4)))  unsigned int  u32x4;

__device__ __forceinline__ v8f wmma_bf16(v16bf a, v16bf b, v8f c) {
  // D = A(16x32 bf16) * B(32x16 bf16) + C(16x16 f32)
  return __builtin_amdgcn_wmma_f32_16x16x32_bf16(false, a, false, b, (short)0, c, false, false);
}

// A-matrix 16x32 bf16 fragment (row-major source, per-lane row = M):
// lanes 0-15: e<8 -> K=k0+e,  e>=8 -> K=k0+e+8
// lanes 16-31: e<8 -> K=k0+e+8, e>=8 -> K=k0+e+16
__device__ __forceinline__ v16bf frag_a(const __bf16* row, int k0, int hi) {
  const __bf16* p = row + k0 + (hi ? 8 : 0);
  union { v16bf v; u32x4 q[2]; } f;
  f.q[0] = *(const u32x4*)(p);
  f.q[1] = *(const u32x4*)(p + 16);
  return f.v;
}

// B-matrix 32x16 bf16 fragment (per-lane column = N, source stores B^T rows):
// lanes 0-15: e<8 -> K=k0+e, e>=8 -> K=k0+e+8; lanes 16-31: e<8 -> K=k0+e+8, e>=8 -> K=k0+e+16
__device__ __forceinline__ v16bf frag_b(const __bf16* row, int k0, int hi) {
  const __bf16* p = row + k0 + (hi ? 8 : 0);
  union { v16bf v; u32x4 q[2]; } f;
  f.q[0] = *(const u32x4*)(p);
  f.q[1] = *(const u32x4*)(p + 16);
  return f.v;
}

__device__ __forceinline__ u32x4 pack8(const float* f) {
  union { __bf16 h[8]; u32x4 u; } r;
#pragma unroll
  for (int i = 0; i < 8; ++i) r.h[i] = (__bf16)f[i];
  return r.u;
}

// ---------------------------------------------------------------------------
// Kernel 1: weights -> bf16, fold attention scale into q/k rows of qkv
// ---------------------------------------------------------------------------
__global__ __launch_bounds__(256) void cvt_weights(
    const float* __restrict__ qw, const float* __restrict__ pw,
    const float* __restrict__ qb,
    __bf16* __restrict__ qwb, __bf16* __restrict__ pwb, float* __restrict__ qbs) {
  const float S = 0.35355339059327373f;  // 1/sqrt(sqrt(64))
  size_t i = (size_t)blockIdx.x * 256 + threadIdx.x;
  if (i < (size_t)L3C * Cc) {
    int row = (int)(i >> 10);
    qwb[i] = (__bf16)(qw[i] * (row < 2 * Cc ? S : 1.f));
  }
  if (i < (size_t)Cc * Cc) pwb[i] = (__bf16)pw[i];
  if (i < L3C) qbs[i] = qb[i] * ((int)i < 2 * Cc ? S : 1.f);
}

// ---------------------------------------------------------------------------
// Kernel 2: GroupNorm.  Writes xn f32 [B][C][L] and xn_t bf16 [B][L][C].
// One block per (b, group): 32 channels x 4096.
// ---------------------------------------------------------------------------
__global__ __launch_bounds__(256) void gn_kernel(
    const float* __restrict__ x, const float* __restrict__ gw,
    const float* __restrict__ gb, float* __restrict__ xn,
    __bf16* __restrict__ xnt) {
  const int b = blockIdx.x >> 5;
  const int g = blockIdx.x & 31;
  const int c0 = g * CPG;
  const int t = threadIdx.x;
  const float* xb = x + ((size_t)b * Cc + c0) * Ll;

  float s = 0.f, ss = 0.f;
  for (int i = t; i < CPG * Ll; i += 256) {
    float v = xb[i];
    s += v; ss += v * v;
  }
  __shared__ float rs[256], rq[256];
  rs[t] = s; rq[t] = ss;
  __syncthreads();
  for (int o = 128; o > 0; o >>= 1) {
    if (t < o) { rs[t] += rs[t + o]; rq[t] += rq[t + o]; }
    __syncthreads();
  }
  const float inv_n = 1.f / (float)(CPG * Ll);
  const float mean = rs[0] * inv_n;
  const float var  = rq[0] * inv_n - mean * mean;
  const float rstd = rsqrtf(var + 1e-5f);

  const int csub = t >> 6;   // 0..3
  const int ll   = t & 63;
  float w8[8], b8[8];
#pragma unroll
  for (int j = 0; j < 8; ++j) {
    w8[j] = gw[c0 + csub + 4 * j];
    b8[j] = gb[c0 + csub + 4 * j];
  }
  __shared__ __align__(16) __bf16 tile[64][32];
  for (int l0 = 0; l0 < Ll; l0 += 64) {
#pragma unroll
    for (int j = 0; j < 8; ++j) {
      int c = csub + 4 * j;
      float v = xb[(size_t)c * Ll + l0 + ll];
      float y = (v - mean) * rstd * w8[j] + b8[j];
      xn[((size_t)b * Cc + c0 + c) * Ll + l0 + ll] = y;
      tile[ll][c] = (__bf16)y;
    }
    __syncthreads();
    int row = t >> 2, seg = t & 3;
    *(u32x4*)&xnt[((size_t)b * Ll + l0 + row) * Cc + c0 + seg * 8] =
        *(const u32x4*)&tile[row][seg * 8];
    __syncthreads();
  }
}

// ---------------------------------------------------------------------------
// Kernel 3: QKV GEMM, 2x2 register-blocked (32x32 per wave).
// out[o,l] = sum_c W[o,c] * xn[c,l] + bias[o]
// q/k rows -> qk bf16 [B][2C][L] (scale pre-folded), v rows -> vt [B*H][L][64]
// ---------------------------------------------------------------------------
__global__ __launch_bounds__(256) void qkv_gemm(
    const __bf16* __restrict__ W, const __bf16* __restrict__ xnt,
    const float* __restrict__ bias, __bf16* __restrict__ qk,
    __bf16* __restrict__ vt) {
  const int b    = blockIdx.z;
  const int n0   = blockIdx.x * 32;
  const int wave = threadIdx.x >> 5;
  const int lane = threadIdx.x & 31;
  const int ln   = lane & 15, hi = lane >> 4;
  const int mt   = blockIdx.y * 256 + wave * 32;

  const __bf16* arow0 = W   + (size_t)(mt + ln) * Cc;
  const __bf16* arow1 = W   + (size_t)(mt + 16 + ln) * Cc;
  const __bf16* brow0 = xnt + ((size_t)b * Ll + n0 + ln) * Cc;
  const __bf16* brow1 = xnt + ((size_t)b * Ll + n0 + 16 + ln) * Cc;

  const v8f z = {0.f, 0.f, 0.f, 0.f, 0.f, 0.f, 0.f, 0.f};
  v8f acc[2][2] = {{z, z}, {z, z}};
#pragma unroll 2
  for (int k = 0; k < Cc; k += 32) {
    v16bf a0 = frag_a(arow0, k, hi);
    v16bf a1 = frag_a(arow1, k, hi);
    v16bf b0 = frag_b(brow0, k, hi);
    v16bf b1 = frag_b(brow1, k, hi);
    acc[0][0] = wmma_bf16(a0, b0, acc[0][0]);
    acc[0][1] = wmma_bf16(a0, b1, acc[0][1]);
    acc[1][0] = wmma_bf16(a1, b0, acc[1][0]);
    acc[1][1] = wmma_bf16(a1, b1, acc[1][1]);
  }

  __shared__ __align__(16) __bf16 st[8][32][32];
  if (mt < 2 * Cc) {  // q or k (block-uniform: 2048 % 256 == 0)
#pragma unroll
    for (int mi = 0; mi < 2; ++mi)
#pragma unroll
      for (int nj = 0; nj < 2; ++nj)
#pragma unroll
        for (int r = 0; r < 8; ++r)
          st[wave][mi * 16 + hi * 8 + r][nj * 16 + ln] =
              (__bf16)(acc[mi][nj][r] + bias[mt + mi * 16 + hi * 8 + r]);
    __syncthreads();
    const int row4 = lane >> 2, seg = lane & 3;
#pragma unroll
    for (int it = 0; it < 4; ++it) {
      int row = row4 + it * 8;
      *(u32x4*)&qk[((size_t)b * (2 * Cc) + mt + row) * Ll + n0 + seg * 8] =
          *(const u32x4*)&st[wave][row][seg * 8];
    }
  } else {            // v: transposed [head][l][ch] store, direct 16B per lane
#pragma unroll
    for (int mi = 0; mi < 2; ++mi) {
      int o2 = mt + mi * 16 - 2 * Cc;
      int hd = o2 >> 6, chb = o2 & 63;
#pragma unroll
      for (int nj = 0; nj < 2; ++nj) {
        float res[8];
#pragma unroll
        for (int r = 0; r < 8; ++r)
          res[r] = acc[mi][nj][r] + bias[mt + mi * 16 + hi * 8 + r];
        *(u32x4*)&vt[(((size_t)b * Hh + hd) * Ll + n0 + nj * 16 + ln) * CHh +
                     chb + hi * 8] = pack8(res);
      }
    }
  }
}

// ---------------------------------------------------------------------------
// Kernel 4: per-head attention.  One block (16 waves) per head.
// score(64x64, K=4096) -> softmax -> ctx(64 x 4096, K=64) -> ctx_t [B][L][C]
// ---------------------------------------------------------------------------
__global__ __launch_bounds__(512) void attn_kernel(
    const __bf16* __restrict__ qk, const __bf16* __restrict__ vt,
    __bf16* __restrict__ ctxt) {
  const int hd   = blockIdx.x;
  const int b    = hd >> 4, h = hd & 15;
  const int wave = threadIdx.x >> 5;
  const int lane = threadIdx.x & 31;
  const int ln   = lane & 15, hi = lane >> 4;

  __shared__ float sc[64][64];
  __shared__ __align__(16) __bf16 wb[64][64];

  {  // score = (q s)(k s)^T, 16 tiles of 16x16, K = L = 4096
    const int mi = wave >> 2, ni = wave & 3;
    const __bf16* qrow = qk + ((size_t)b * (2 * Cc) + h * CHh + mi * 16 + ln) * Ll;
    const __bf16* krow = qk + ((size_t)b * (2 * Cc) + Cc + h * CHh + ni * 16 + ln) * Ll;
    v8f acc = {0.f, 0.f, 0.f, 0.f, 0.f, 0.f, 0.f, 0.f};
#pragma unroll 4
    for (int k = 0; k < Ll; k += 32)
      acc = wmma_bf16(frag_a(qrow, k, hi), frag_b(krow, k, hi), acc);
#pragma unroll
    for (int r = 0; r < 8; ++r) sc[mi * 16 + hi * 8 + r][ni * 16 + ln] = acc[r];
  }
  __syncthreads();
  if (threadIdx.x < 64) {  // row softmax in f32, result to bf16 LDS
    const int row = threadIdx.x;
    float mx = -3.4e38f;
    for (int d = 0; d < 64; ++d) mx = fmaxf(mx, sc[row][d]);
    float sum = 0.f;
    for (int d = 0; d < 64; ++d) {
      float e = __expf(sc[row][d] - mx);
      sum += e; sc[row][d] = e;
    }
    float inv = 1.f / sum;
    for (int d = 0; d < 64; ++d) wb[row][d] = (__bf16)(sc[row][d] * inv);
  }
  __syncthreads();
  {  // ctx = W @ v : M=64 (A cached from LDS), N=4096, K=64 (2 WMMAs/tile)
    const int mi = wave & 3;
    const __bf16* arow = &wb[mi * 16 + ln][0];
    v16bf a0 = frag_a(arow, 0, hi);
    v16bf a1 = frag_a(arow, 32, hi);
    const __bf16* vbase = vt + ((size_t)b * Hh + h) * (size_t)Ll * CHh;
    for (int nt = (wave >> 2); nt < Ll / 16; nt += 4) {
      const __bf16* brow = vbase + (size_t)(nt * 16 + ln) * CHh;
      v8f acc = {0.f, 0.f, 0.f, 0.f, 0.f, 0.f, 0.f, 0.f};
      acc = wmma_bf16(a0, frag_b(brow, 0, hi), acc);
      acc = wmma_bf16(a1, frag_b(brow, 32, hi), acc);
      float res[8];
#pragma unroll
      for (int r = 0; r < 8; ++r) res[r] = acc[r];
      *(u32x4*)&ctxt[((size_t)b * Ll + nt * 16 + ln) * Cc + h * CHh + mi * 16 + hi * 8] =
          pack8(res);
    }
  }
}

// ---------------------------------------------------------------------------
// Kernel 5: proj GEMM (2x2 blocked) + bias + residual.  out = xn + (P@ctx + pb)
// ---------------------------------------------------------------------------
__global__ __launch_bounds__(256) void proj_gemm(
    const __bf16* __restrict__ P, const __bf16* __restrict__ ctxt,
    const float* __restrict__ pb, const float* __restrict__ xn,
    float* __restrict__ out) {
  const int b    = blockIdx.z;
  const int n0   = blockIdx.x * 32;
  const int wave = threadIdx.x >> 5;
  const int lane = threadIdx.x & 31;
  const int ln   = lane & 15, hi = lane >> 4;
  const int mt   = blockIdx.y * 256 + wave * 32;

  const __bf16* arow0 = P    + (size_t)(mt + ln) * Cc;
  const __bf16* arow1 = P    + (size_t)(mt + 16 + ln) * Cc;
  const __bf16* brow0 = ctxt + ((size_t)b * Ll + n0 + ln) * Cc;
  const __bf16* brow1 = ctxt + ((size_t)b * Ll + n0 + 16 + ln) * Cc;

  const v8f z = {0.f, 0.f, 0.f, 0.f, 0.f, 0.f, 0.f, 0.f};
  v8f acc[2][2] = {{z, z}, {z, z}};
#pragma unroll 2
  for (int k = 0; k < Cc; k += 32) {
    v16bf a0 = frag_a(arow0, k, hi);
    v16bf a1 = frag_a(arow1, k, hi);
    v16bf b0 = frag_b(brow0, k, hi);
    v16bf b1 = frag_b(brow1, k, hi);
    acc[0][0] = wmma_bf16(a0, b0, acc[0][0]);
    acc[0][1] = wmma_bf16(a0, b1, acc[0][1]);
    acc[1][0] = wmma_bf16(a1, b0, acc[1][0]);
    acc[1][1] = wmma_bf16(a1, b1, acc[1][1]);
  }

  __shared__ float st[8][32][32];  // 32 KB
#pragma unroll
  for (int mi = 0; mi < 2; ++mi)
#pragma unroll
    for (int nj = 0; nj < 2; ++nj)
#pragma unroll
      for (int r = 0; r < 8; ++r)
        st[wave][mi * 16 + hi * 8 + r][nj * 16 + ln] =
            acc[mi][nj][r] + pb[mt + mi * 16 + hi * 8 + r];
  __syncthreads();
  const int row4 = lane >> 2, seg = lane & 3;
#pragma unroll
  for (int it = 0; it < 4; ++it) {
    int row = row4 + it * 8;
    size_t off = ((size_t)b * Cc + mt + row) * Ll + n0 + seg * 8;
#pragma unroll
    for (int i = 0; i < 8; ++i) out[off + i] = xn[off + i] + st[wave][row][seg * 8 + i];
  }
}

// ---------------------------------------------------------------------------
extern "C" void kernel_launch(void* const* d_in, const int* in_sizes, int n_in,
                              void* d_out, int out_size, void* d_ws, size_t ws_size,
                              hipStream_t stream) {
  (void)in_sizes; (void)n_in; (void)out_size; (void)ws_size;
  const float* x      = (const float*)d_in[0];
  const float* gn_w   = (const float*)d_in[1];
  const float* gn_b   = (const float*)d_in[2];
  const float* qkv_w  = (const float*)d_in[3];
  const float* qkv_b  = (const float*)d_in[4];
  const float* proj_w = (const float*)d_in[5];
  const float* proj_b = (const float*)d_in[6];
  float* out = (float*)d_out;

  char* ws = (char*)d_ws;
  float*  xn     = (float*)ws;   ws += (size_t)Bz * Cc * Ll * 4;        // 134 MB
  __bf16* xnt    = (__bf16*)ws;  ws += (size_t)Bz * Ll * Cc * 2;        //  67 MB
  __bf16* qk     = (__bf16*)ws;  ws += (size_t)Bz * 2 * Cc * Ll * 2;    // 134 MB
  __bf16* vt     = (__bf16*)ws;  ws += (size_t)Bz * Hh * Ll * CHh * 2;  //  67 MB
  __bf16* ctxt   = (__bf16*)ws;  ws += (size_t)Bz * Ll * Cc * 2;        //  67 MB
  __bf16* qkvw_b = (__bf16*)ws;  ws += (size_t)L3C * Cc * 2;            //   6 MB
  __bf16* projw_b= (__bf16*)ws;  ws += (size_t)Cc * Cc * 2;             //   2 MB
  float*  qkvb_s = (float*)ws;   ws += (size_t)L3C * 4;

  cvt_weights<<<(L3C * Cc + 255) / 256, 256, 0, stream>>>(
      qkv_w, proj_w, qkv_b, qkvw_b, projw_b, qkvb_s);
  gn_kernel<<<Bz * 32, 256, 0, stream>>>(x, gn_w, gn_b, xn, xnt);
  qkv_gemm<<<dim3(Ll / 32, L3C / 256, Bz), 256, 0, stream>>>(
      qkvw_b, xnt, qkvb_s, qk, vt);
  attn_kernel<<<Bz * Hh, 512, 0, stream>>>(qk, vt, ctxt);
  proj_gemm<<<dim3(Ll / 32, Cc / 256, Bz), 256, 0, stream>>>(
      projw_b, ctxt, proj_b, xn, out);
}
